// GraphConv_15195594293935
// MI455X (gfx1250) — matrix operand order repaired and checked by reading, not compiled, not run on hardware
//
#include <hip/hip_runtime.h>

typedef __bf16 bf16_t;
typedef bf16_t v16bf __attribute__((ext_vector_type(16)));
typedef bf16_t v8bf  __attribute__((ext_vector_type(8)));
typedef float  v8f   __attribute__((ext_vector_type(8)));

#define N_NODES 20000
#define N_EDGES 160000
#define N_IN    512
#define N_OUT   512

// Workspace layout (bytes)
#define AGG_F32_OFF   0u
#define AGG_BF16_OFF  40960000u   // 20000*512*4
#define W_BF16_OFF    61440000u   // + 20000*512*2

// ---------------------------------------------------------------------------
// Phase 1: zero the fp32 accumulator (one float4 per thread)
// ---------------------------------------------------------------------------
__global__ void __launch_bounds__(256) gc_zero_f4(float4* __restrict__ p, int n4) {
    int i = blockIdx.x * 256 + threadIdx.x;
    if (i < n4) p[i] = make_float4(0.f, 0.f, 0.f, 0.f);
}

// ---------------------------------------------------------------------------
// Phase 2: edge scatter  agg[row[e]] += vals[e] * x[col[e]]
// One wave32 per edge: 512 floats = 4 float4 per lane, coalesced row read,
// fp32 atomic adds resolving in L2 (agg 41MB << 192MB L2).
// ---------------------------------------------------------------------------
__global__ void __launch_bounds__(256) gc_edge_scatter(
    const float* __restrict__ x, const float* __restrict__ vals,
    const int* __restrict__ row, const int* __restrict__ col,
    float* __restrict__ agg) {
    const int lane = threadIdx.x & 31;
    const int wave = threadIdx.x >> 5;
    const int e = blockIdx.x * 8 + wave;          // grid = E/8 blocks

    const float v = vals[e];
    const int   r = row[e];
    const int   c = col[e];

    const float4* __restrict__ xp = (const float4*)(x + (size_t)c * N_IN);
    float* __restrict__ ag = agg + (size_t)r * N_IN;

#pragma unroll
    for (int j = 0; j < 4; ++j) {
        float4 d = xp[lane + 32 * j];
        int b = 4 * (lane + 32 * j);
        atomicAdd(ag + b + 0, v * d.x);
        atomicAdd(ag + b + 1, v * d.y);
        atomicAdd(ag + b + 2, v * d.z);
        atomicAdd(ag + b + 3, v * d.w);
    }
}

// ---------------------------------------------------------------------------
// Phase 3a: streaming fp32 -> bf16 conversion (8 elems/thread: 32B in, 16B out)
// ---------------------------------------------------------------------------
__global__ void __launch_bounds__(256) gc_cvt_bf16(
    const float* __restrict__ src, bf16_t* __restrict__ dst, int n8) {
    int i = blockIdx.x * 256 + threadIdx.x;
    if (i >= n8) return;
    const float4* s = (const float4*)src + 2 * (size_t)i;
    float4 a = s[0], b = s[1];
    v8bf o;
    o[0] = (bf16_t)a.x; o[1] = (bf16_t)a.y; o[2] = (bf16_t)a.z; o[3] = (bf16_t)a.w;
    o[4] = (bf16_t)b.x; o[5] = (bf16_t)b.y; o[6] = (bf16_t)b.z; o[7] = (bf16_t)b.w;
    *(v8bf*)(dst + 8 * (size_t)i) = o;
}

// ---------------------------------------------------------------------------
// Phase 3b: out = agg @ W^T via V_WMMA_F32_16X16X32_BF16, bf16 operands.
// Block = 8 waves; block owns one 16-row M tile; wave w owns columns
// [64w, 64w+64) as four 16x16 N tiles, A fragment reused across the 4 WMMAs.
//
// Fragment layouts used here (contiguous k, not the table order of CDNA5 ISA §7.12.2):
//   A 16x32 bf16 : lane L -> row L&15 ; ksel = 16*(L>=16);
//                  elems 0..15 = K ksel .. ksel+15
//   B 32x16 bf16 : lane L -> col L&15 ; elems 0..15 = K 16*(L>=16) .. +15
//   C/D 16x16 f32: VGPR v -> row v + 8*(L>=16), lane -> col L&15
// ---------------------------------------------------------------------------
__global__ void __launch_bounds__(256) gc_gemm_wmma(
    const bf16_t* __restrict__ A, const bf16_t* __restrict__ Wb,
    float* __restrict__ out) {
    const int lane  = threadIdx.x & 31;
    const int wave  = threadIdx.x >> 5;       // 0..7
    const int mBase = blockIdx.x * 16;        // 1250 blocks
    const int nBase = wave * 64;

    const int aRow  = mBase + (lane & 15);
    const int aKsel = (lane >> 4) << 4;       // 0 or 16
    const int bN    = lane & 15;
    const int bKsel = (lane >> 4) << 4;       // 0 or 16

    const bf16_t* __restrict__ ap = A + (size_t)aRow * N_IN + aKsel;
    const bf16_t* __restrict__ w0 = Wb + (size_t)(nBase +  0 + bN) * N_IN + bKsel;
    const bf16_t* __restrict__ w1 = Wb + (size_t)(nBase + 16 + bN) * N_IN + bKsel;
    const bf16_t* __restrict__ w2 = Wb + (size_t)(nBase + 32 + bN) * N_IN + bKsel;
    const bf16_t* __restrict__ w3 = Wb + (size_t)(nBase + 48 + bN) * N_IN + bKsel;

    v8f c0 = {}, c1 = {}, c2 = {}, c3 = {};

#pragma unroll 2
    for (int k = 0; k < N_IN; k += 32) {
        v8bf alo = *(const v8bf*)(ap + k);
        v8bf ahi = *(const v8bf*)(ap + k + 8);
        v16bf a = __builtin_shufflevector(alo, ahi,
            0, 1, 2, 3, 4, 5, 6, 7, 8, 9, 10, 11, 12, 13, 14, 15);
        v16bf b0 = *(const v16bf*)(w0 + k);
        v16bf b1 = *(const v16bf*)(w1 + k);
        v16bf b2 = *(const v16bf*)(w2 + k);
        v16bf b3 = *(const v16bf*)(w3 + k);
        c0 = __builtin_amdgcn_wmma_f32_16x16x32_bf16(false, a, false, b0, (short)0, c0, false, false);
        c1 = __builtin_amdgcn_wmma_f32_16x16x32_bf16(false, a, false, b1, (short)0, c1, false, false);
        c2 = __builtin_amdgcn_wmma_f32_16x16x32_bf16(false, a, false, b2, (short)0, c2, false, false);
        c3 = __builtin_amdgcn_wmma_f32_16x16x32_bf16(false, a, false, b3, (short)0, c3, false, false);
    }

    const int mOff = (lane >> 4) << 3;        // 0 or 8
    const int nCol = lane & 15;
    float* __restrict__ op = out + (size_t)(mBase + mOff) * N_OUT + nBase + nCol;
#pragma unroll
    for (int v = 0; v < 8; ++v) {
        op[(size_t)v * N_OUT +  0] = c0[v];
        op[(size_t)v * N_OUT + 16] = c1[v];
        op[(size_t)v * N_OUT + 32] = c2[v];
        op[(size_t)v * N_OUT + 48] = c3[v];
    }
}

// ---------------------------------------------------------------------------
extern "C" void kernel_launch(void* const* d_in, const int* in_sizes, int n_in,
                              void* d_out, int out_size, void* d_ws, size_t ws_size,
                              hipStream_t stream) {
    const float* x      = (const float*)d_in[0];
    const float* vals   = (const float*)d_in[1];
    const float* weight = (const float*)d_in[2];
    const int*   row    = (const int*)d_in[3];
    const int*   col    = (const int*)d_in[4];
    float*       out    = (float*)d_out;

    char* ws = (char*)d_ws;
    float*  agg  = (float*)(ws + AGG_F32_OFF);    // 20000*512 fp32
    bf16_t* aggb = (bf16_t*)(ws + AGG_BF16_OFF);  // 20000*512 bf16
    bf16_t* wb   = (bf16_t*)(ws + W_BF16_OFF);    // 512*512 bf16

    const int n4 = N_NODES * N_IN / 4;            // 2,560,000 float4
    gc_zero_f4<<<(n4 + 255) / 256, 256, 0, stream>>>((float4*)agg, n4);

    gc_edge_scatter<<<N_EDGES / 8, 256, 0, stream>>>(x, vals, row, col, agg);

    const int nAgg8 = N_NODES * N_IN / 8;         // 1,280,000
    const int nW8   = N_OUT * N_IN / 8;           // 32,768
    gc_cvt_bf16<<<(nAgg8 + 255) / 256, 256, 0, stream>>>(agg, aggb, nAgg8);
    gc_cvt_bf16<<<(nW8 + 255) / 256, 256, 0, stream>>>(weight, wb, nW8);

    gc_gemm_wmma<<<N_NODES / 16, 256, 0, stream>>>(aggb, wb, out);
}
